// SSingleMatchNet_69552700391693
// MI455X (gfx1250) — matrix operand order repaired and checked by reading, not compiled, not run on hardware
//
#include <hip/hip_runtime.h>
#include <hip/hip_bf16.h>
#include <cstdint>

// ---------------------------------------------------------------------------
// SSingleMatchNet forward for MI455X (gfx1250, wave32, WMMA bf16)
//
//   B=8, LP=2048, LQ=2048, H=1024
//   tq  = q @ W^T + b                (GEMM1, bf16 WMMA, K=H contiguous both sides)
//   S   = p @ tq^T ; att=softmax(S)  (GEMM2 fused with masked softmax, rows in regs)
//   out = att @ q                    (GEMM3, B operand pre-transposed so K contiguous)
//
// All matrix math: v_wmma_f32_16x16x32_bf16 (16384 FLOP/instr). Fragments are
// loaded straight from global (L2 = 192MB holds every reused operand), with
// explicit software pipelining so ~8 global_load_b128 are always in flight
// behind the WMMA stream.
// ---------------------------------------------------------------------------

#define BATCH 8
#define LPD   2048
#define LQD   2048
#define HD    1024

typedef __bf16 bf16_t;
typedef bf16_t v16bf __attribute__((ext_vector_type(16)));
typedef bf16_t v8bf  __attribute__((ext_vector_type(8)));
typedef bf16_t v4bf  __attribute__((ext_vector_type(4)));
typedef float  v8f   __attribute__((ext_vector_type(8)));

__device__ __forceinline__ bf16_t f2bf(float f) {
  unsigned u = __builtin_bit_cast(unsigned, f);
  u += 0x7fffu + ((u >> 16) & 1u);              // round-to-nearest-even
  unsigned short h = (unsigned short)(u >> 16);
  return __builtin_bit_cast(bf16_t, h);
}

// A-fragment (16x32 bf16, M-row per lane): two 16B loads at +0 and +8 halves (16 contiguous K).
__device__ __forceinline__ v16bf load_a_frag(const bf16_t* p) {
  v8bf lo = *reinterpret_cast<const v8bf*>(p);
  v8bf hi = *reinterpret_cast<const v8bf*>(p + 8);
  return __builtin_shufflevector(lo, hi, 0,1,2,3,4,5,6,7,8,9,10,11,12,13,14,15);
}

__device__ __forceinline__ v8f wmma_bf16(v16bf a, v16bf b, v8f c) {
  return __builtin_amdgcn_wmma_f32_16x16x32_bf16(false, a, false, b, (short)0, c,
                                                 false, false);
}

// ---------------------------------------------------------------------------
// K1a: flat f32 -> bf16 conversion (vectorized x4)
// ---------------------------------------------------------------------------
__global__ void __launch_bounds__(256) cvt_f32_bf16_kernel(
    const float* __restrict__ in, bf16_t* __restrict__ out, int n) {
  int i = (blockIdx.x * blockDim.x + threadIdx.x) * 4;
  if (i + 3 < n) {
    float4 v = *reinterpret_cast<const float4*>(in + i);
    v4bf s = {f2bf(v.x), f2bf(v.y), f2bf(v.z), f2bf(v.w)};
    *reinterpret_cast<v4bf*>(out + i) = s;
  } else {
    for (int j = i; j < n; ++j) out[j] = f2bf(in[j]);
  }
}

// ---------------------------------------------------------------------------
// K1b: proj_q [B][LQ][H] f32  ->  qT16 [B][H][LQ] bf16  (64x64 LDS tile)
// ---------------------------------------------------------------------------
__global__ void __launch_bounds__(256) transpose_cvt_kernel(
    const float* __restrict__ q, bf16_t* __restrict__ qT) {
  __shared__ bf16_t tile[64][65];
  int b  = blockIdx.z;
  int h0 = blockIdx.x * 64;
  int q0 = blockIdx.y * 64;
  int tx = threadIdx.x & 63;
  int ty = threadIdx.x >> 6;  // 0..3
  const float* src = q + ((size_t)b * LQD + q0) * HD + h0;
#pragma unroll
  for (int i = 0; i < 16; ++i) {
    int qr = ty + i * 4;
    tile[qr][tx] = f2bf(src[(size_t)qr * HD + tx]);
  }
  __syncthreads();
  bf16_t* dst = qT + ((size_t)b * HD + h0) * LQD + q0;
#pragma unroll
  for (int i = 0; i < 16; ++i) {
    int hr = ty + i * 4;
    dst[(size_t)hr * LQD + tx] = tile[tx][hr];
  }
}

// ---------------------------------------------------------------------------
// Generic  C[M][N] = A(MxK, row-major) * B(NxK, row-major)^T
// Block = 256 threads = 8 waves in 4(M) x 2(N); wave tile 32x64 (2x4 WMMA)
// -> block tile 128x128; 1.0 global loads per WMMA in the main loop.
// Software-pipelined: the whole next-k operand set (2 A frags + 4 B frags)
// is fetched before the 8 WMMAs of the current step are issued.
// MODE 0: f32 out.  MODE 1: += bias[N], bf16 out.
// ---------------------------------------------------------------------------
template <int MODE>
__global__ void __launch_bounds__(256, 1) gemm_rt_kernel(
    const bf16_t* __restrict__ A, long strideAb, int lda,
    const bf16_t* __restrict__ Bm, long strideBb, int ldb,
    int K,
    void* __restrict__ outp, long strideOb, int ldo,
    const float* __restrict__ bias) {
  int b    = blockIdx.z;
  const bf16_t* Ab = A + (size_t)b * strideAb;
  const bf16_t* Bb = Bm + (size_t)b * strideBb;
  int wave = threadIdx.x >> 5;
  int lane = threadIdx.x & 31;
  int lr   = lane & 15;
  int hi   = (lane >> 4) & 1;
  int m0   = blockIdx.x * 128 + (wave >> 1) * 32;   // 4 waves along M
  int n0   = blockIdx.y * 128 + (wave & 1) * 64;    // 2 waves along N

  int ka = hi ? 16 : 0;   // A element K-base per half-wave
  int kb = hi ? 16 : 0;   // B element K-base per half-wave

  const bf16_t* arow0 = Ab + (size_t)(m0 + lr) * lda + ka;
  const bf16_t* arow1 = arow0 + (size_t)16 * lda;
  const bf16_t* brow  = Bb + (size_t)(n0 + lr) * ldb + kb;  // + tn*16*ldb via imm

  v8f acc[2][4] = {};

  // pipeline prologue
  v16bf a0c = load_a_frag(arow0);
  v16bf a1c = load_a_frag(arow1);
  v16bf bc[4];
#pragma unroll
  for (int tn = 0; tn < 4; ++tn)
    bc[tn] = *reinterpret_cast<const v16bf*>(brow + (size_t)tn * 16 * ldb);

#pragma unroll 1
  for (int kk = 0; kk < K; kk += 32) {
    int kn = (kk + 32 < K) ? kk + 32 : kk;     // clamp: stay in bounds
    __builtin_prefetch(arow0 + kk + 512, 0, 0);  // global_prefetch_b8 (L2 warm)
    v16bf a0n = load_a_frag(arow0 + kn);
    v16bf a1n = load_a_frag(arow1 + kn);
    v16bf bn[4];
#pragma unroll
    for (int tn = 0; tn < 4; ++tn)
      bn[tn] = *reinterpret_cast<const v16bf*>(brow + (size_t)tn * 16 * ldb + kn);

#pragma unroll
    for (int tn = 0; tn < 4; ++tn) {
      acc[0][tn] = wmma_bf16(a0c, bc[tn], acc[0][tn]);
      acc[1][tn] = wmma_bf16(a1c, bc[tn], acc[1][tn]);
    }
    a0c = a0n; a1c = a1n;
#pragma unroll
    for (int tn = 0; tn < 4; ++tn) bc[tn] = bn[tn];
  }

#pragma unroll
  for (int tm = 0; tm < 2; ++tm)
#pragma unroll
    for (int tn = 0; tn < 4; ++tn)
#pragma unroll
      for (int r = 0; r < 8; ++r) {
        int M = m0 + tm * 16 + r + hi * 8;
        int N = n0 + tn * 16 + lr;
        float v = acc[tm][tn][r];
        if constexpr (MODE == 1) {
          v += bias[N];
          bf16_t* o = (bf16_t*)outp + (size_t)b * strideOb;
          o[(size_t)M * ldo + N] = f2bf(v);
        } else {
          float* o = (float*)outp + (size_t)b * strideOb;
          o[(size_t)M * ldo + N] = v;
        }
      }
}

// ---------------------------------------------------------------------------
// Fused scores + masked softmax:
//   block = 16 P rows; 8 waves each own 256 contiguous Q columns.
//   S rows (16 x 2048) live entirely in registers (16 accum tiles / wave),
//   then half-wave shuffle + LDS cross-wave reduction for max / sum,
//   att written to global as bf16.  Mask col >= seq_len[b]  (the reference's
//   mask/softmax/re-mask/renorm cancels to a masked softmax over valid keys;
//   seq_len >= 1 so every row has a valid column).
//   B fragments flow through a 4-slot rotating queue: 8 loads in flight
//   behind every WMMA; at t>=12 the queue refills with the NEXT k-step's
//   tiles so the pipeline never drains.
// ---------------------------------------------------------------------------
__global__ void __launch_bounds__(256, 1) scores_softmax_kernel(
    const bf16_t* __restrict__ p16, const bf16_t* __restrict__ tq16,
    bf16_t* __restrict__ att16, const int* __restrict__ seq_len) {
  int bidx = blockIdx.x;
  int b    = bidx >> 7;           // LP/16 = 128 blocks per batch
  int p0   = (bidx & 127) << 4;
  int wave = threadIdx.x >> 5;
  int lane = threadIdx.x & 31;
  int lr   = lane & 15;
  int hi   = (lane >> 4) & 1;
  int nb   = wave * 256;

  int ka = hi ? 16 : 0;
  int kb = hi ? 16 : 0;
  const bf16_t* arow  = p16  + ((size_t)b * LPD + p0 + lr) * HD + ka;
  const bf16_t* bbase = tq16 + ((size_t)b * LQD + nb + lr) * HD + kb;

  v8f acc[16] = {};

  // pipeline prologue: A frag + first 4 B tiles
  v16bf a = load_a_frag(arow);
  v16bf an;
  v16bf bq[4];
#pragma unroll
  for (int t = 0; t < 4; ++t)
    bq[t] = *reinterpret_cast<const v16bf*>(bbase + (size_t)t * 16 * HD);

#pragma unroll 1
  for (int kk = 0; kk < HD; kk += 32) {
    int kn = (kk + 32 < HD) ? kk + 32 : kk;   // clamp: stay in bounds
#pragma unroll
    for (int t = 0; t < 16; ++t) {
      v16bf cur = bq[t & 3];
      if (t < 12)
        bq[t & 3] = *reinterpret_cast<const v16bf*>(bbase + (size_t)(t + 4) * 16 * HD + kk);
      else
        bq[t & 3] = *reinterpret_cast<const v16bf*>(bbase + (size_t)(t - 12) * 16 * HD + kn);
      if (t == 8) an = load_a_frag(arow + kn);
      acc[t] = wmma_bf16(a, cur, acc[t]);
    }
    a = an;
  }

  int sl = seq_len[b];

  // ---- row max (masked) ----
  float rmax[8];
#pragma unroll
  for (int r = 0; r < 8; ++r) rmax[r] = -3.4e38f;
#pragma unroll
  for (int t = 0; t < 16; ++t) {
    int col = nb + t * 16 + lr;
    if (col < sl) {
#pragma unroll
      for (int r = 0; r < 8; ++r) rmax[r] = fmaxf(rmax[r], acc[t][r]);
    }
  }
#pragma unroll
  for (int off = 1; off < 16; off <<= 1)
#pragma unroll
    for (int r = 0; r < 8; ++r)
      rmax[r] = fmaxf(rmax[r], __shfl_xor(rmax[r], off, 32));

  __shared__ float red[8][16];
  if (lane == 0) {
#pragma unroll
    for (int r = 0; r < 8; ++r) red[wave][r] = rmax[r];
  } else if (lane == 16) {
#pragma unroll
    for (int r = 0; r < 8; ++r) red[wave][8 + r] = rmax[r];
  }
  __syncthreads();
  float gmax[8];
#pragma unroll
  for (int r = 0; r < 8; ++r) {
    int M = r + hi * 8;
    float m = -3.4e38f;
#pragma unroll
    for (int w = 0; w < 8; ++w) m = fmaxf(m, red[w][M]);
    gmax[r] = m;
  }
  __syncthreads();

  // ---- exp + row sum ----
  float rsum[8];
#pragma unroll
  for (int r = 0; r < 8; ++r) rsum[r] = 0.f;
#pragma unroll
  for (int t = 0; t < 16; ++t) {
    int col = nb + t * 16 + lr;
    bool valid = col < sl;
#pragma unroll
    for (int r = 0; r < 8; ++r) {
      float e = valid ? __expf(acc[t][r] - gmax[r]) : 0.f;
      acc[t][r] = e;
      rsum[r] += e;
    }
  }
#pragma unroll
  for (int off = 1; off < 16; off <<= 1)
#pragma unroll
    for (int r = 0; r < 8; ++r) rsum[r] += __shfl_xor(rsum[r], off, 32);
  if (lane == 0) {
#pragma unroll
    for (int r = 0; r < 8; ++r) red[wave][r] = rsum[r];
  } else if (lane == 16) {
#pragma unroll
    for (int r = 0; r < 8; ++r) red[wave][8 + r] = rsum[r];
  }
  __syncthreads();
  float ginv[8];
#pragma unroll
  for (int r = 0; r < 8; ++r) {
    int M = r + hi * 8;
    float s = 1e-13f;
#pragma unroll
    for (int w = 0; w < 8; ++w) s += red[w][M];
    ginv[r] = 1.0f / s;
  }

  // ---- normalize + store bf16 ----
  bf16_t* orow = att16 + ((size_t)b * LPD + p0) * LQD;
#pragma unroll
  for (int t = 0; t < 16; ++t) {
    int col = nb + t * 16 + lr;
#pragma unroll
    for (int r = 0; r < 8; ++r) {
      int M = r + hi * 8;
      orow[(size_t)M * LQD + col] = f2bf(acc[t][r] * ginv[r]);
    }
  }
}

// ---------------------------------------------------------------------------
// Launch.  Workspace layout (bytes):
//   [0,32M)    p16     B*LP*H bf16
//   [32M,64M)  q16     B*LQ*H bf16
//   [64M,96M)  qT16    B*H*LQ bf16
//   [96M,128M) tq16    B*LQ*H bf16
//   [128M,130M)W16     H*H    bf16
//   [130M,194M)att16   B*LP*LQ bf16
// Total ~194 MB.
// ---------------------------------------------------------------------------
extern "C" void kernel_launch(void* const* d_in, const int* in_sizes, int n_in,
                              void* d_out, int out_size, void* d_ws, size_t ws_size,
                              hipStream_t stream) {
  const float* proj_p  = (const float*)d_in[0];
  const float* proj_q  = (const float*)d_in[1];
  const float* W_t     = (const float*)d_in[2];
  const float* b_t     = (const float*)d_in[3];
  const int*   seq_len = (const int*)d_in[4];
  float* out = (float*)d_out;

  char* ws = (char*)d_ws;
  const size_t MB = (size_t)1 << 20;
  bf16_t* p16   = (bf16_t*)(ws + 0 * MB);
  bf16_t* q16   = (bf16_t*)(ws + 32 * MB);
  bf16_t* qT16  = (bf16_t*)(ws + 64 * MB);
  bf16_t* tq16  = (bf16_t*)(ws + 96 * MB);
  bf16_t* W16   = (bf16_t*)(ws + 128 * MB);
  bf16_t* att16 = (bf16_t*)(ws + 130 * MB);

  const int nPQ = BATCH * LPD * HD;  // 16M elements
  cvt_f32_bf16_kernel<<<nPQ / 4 / 256, 256, 0, stream>>>(proj_p, p16, nPQ);
  cvt_f32_bf16_kernel<<<nPQ / 4 / 256, 256, 0, stream>>>(proj_q, q16, nPQ);
  cvt_f32_bf16_kernel<<<(HD * HD) / 4 / 256, 256, 0, stream>>>(W_t, W16, HD * HD);
  transpose_cvt_kernel<<<dim3(HD / 64, LQD / 64, BATCH), 256, 0, stream>>>(proj_q, qT16);

  // GEMM1: tq16[b] = q16[b] (LQ x H) * W16^T (H x H) + b_t   -> bf16
  gemm_rt_kernel<1><<<dim3(LQD / 128, HD / 128, BATCH), 256, 0, stream>>>(
      q16, (long)LQD * HD, HD, W16, 0L, HD, HD, tq16, (long)LQD * HD, HD, b_t);

  // GEMM2 + masked softmax fused
  scores_softmax_kernel<<<dim3(BATCH * (LPD / 16)), 256, 0, stream>>>(
      p16, tq16, att16, seq_len);

  // GEMM3: out[b] = att16[b] (LP x LQ) * qT16[b]^T (LQ x H)  -> f32
  gemm_rt_kernel<0><<<dim3(LPD / 128, HD / 128, BATCH), 256, 0, stream>>>(
      att16, (long)LPD * LQD, LQD, qT16, (long)HD * LQD, LQD, LQD,
      out, (long)LPD * HD, HD, nullptr);
}
